// Net_37958920962591
// MI455X (gfx1250) — compile-verified, matrix operand order checked by reading, not run on hardware
//
#include <hip/hip_runtime.h>

#define HD __device__ __forceinline__

typedef __attribute__((ext_vector_type(16))) __bf16 v16bf;
typedef __attribute__((ext_vector_type(8)))  float  v8f;

// float -> bf16 bits, round-to-nearest-even
HD unsigned short f2bf_bits(float f) {
  unsigned u = __builtin_bit_cast(unsigned, f);
  unsigned r = u + 0x7FFFu + ((u >> 16) & 1u);
  return (unsigned short)(r >> 16);
}
HD unsigned long long pack4_bf16(float4 v) {
  unsigned long long a = f2bf_bits(v.x);
  unsigned long long b = f2bf_bits(v.y);
  unsigned long long c = f2bf_bits(v.z);
  unsigned long long d = f2bf_bits(v.w);
  return a | (b << 16) | (c << 32) | (d << 48);
}
HD float sigm(float x) { return 1.0f / (1.0f + __expf(-x)); }

// 16x32 bf16 A/B fragment (wave32 layout per CDNA5 ISA 7.12.2):
// lanes 0-15: row=lane, K = 0..7 / 16..23 ; lanes 16-31: row=lane-16, K = 8..15 / 24..31
HD v16bf frag_from_lds(const unsigned short* row_base, int stride, int lane) {
  int m = lane & 15, half = lane >> 4;
  const unsigned short* p = row_base + m * stride;
  v16bf f;
#pragma unroll
  for (int e = 0; e < 16; ++e) {
    int k = (e < 8) ? (half * 8 + e) : (16 + half * 8 + (e - 8));
    f[e] = __builtin_bit_cast(__bf16, p[k]);
  }
  return f;
}

// ---------------------------------------------------------------------------
// GEMM: out = act( A[M,K] @ W[N,K]^T + bias1 + bias2 ), bf16 WMMA, f32 acc.
// Block tile 64(M) x 128(N); 8 waves in a 2x4 grid; each wave: 2x2 WMMA tiles.
// Requires M%64==0, N%128==0, K%32==0 (all shapes here comply -> no guards).
// trans==1: input row r = b*Tdim + t -> output row t*Bdim + b ([B,T] -> [T,B]).
// ---------------------------------------------------------------------------
__global__ __launch_bounds__(256)
void gemm_bf16_wmma(const float* __restrict__ A, const float* __restrict__ W,
                    const float* __restrict__ bias1, const float* __restrict__ bias2,
                    float* __restrict__ out, int M, int K, int N,
                    int act, int trans, int Tdim, int Bdim) {
  __shared__ unsigned short sA[64 * 32];
  __shared__ unsigned short sB[128 * 32];
  int tid = threadIdx.x;
  int lane = tid & 31, w = tid >> 5;
  int mh2 = (w >> 2) * 32;                   // wave's 32-row slice of A tile
  int nw2 = (w & 3) * 32;                    // wave's 32-col slice of B tile
  int nBase = blockIdx.x * 128;
  int mBase = blockIdx.y * 64;
  v8f acc00 = {}, acc01 = {}, acc10 = {}, acc11 = {};
  for (int kk = 0; kk < K; kk += 32) {
    // vectorized staging: float4 global loads, packed 4xbf16 b64 LDS stores
    for (int i = tid; i < 64 * 8; i += 256) {
      int r = i >> 3, c4 = (i & 7) * 4;
      float4 v = *reinterpret_cast<const float4*>(A + (size_t)(mBase + r) * K + kk + c4);
      *reinterpret_cast<unsigned long long*>(sA + r * 32 + c4) = pack4_bf16(v);
    }
    for (int i = tid; i < 128 * 8; i += 256) {
      int r = i >> 3, c4 = (i & 7) * 4;
      float4 v = *reinterpret_cast<const float4*>(W + (size_t)(nBase + r) * K + kk + c4);
      *reinterpret_cast<unsigned long long*>(sB + r * 32 + c4) = pack4_bf16(v);
    }
    __syncthreads();
    v16bf a0 = frag_from_lds(sA + (mh2) * 32, 32, lane);
    v16bf a1 = frag_from_lds(sA + (mh2 + 16) * 32, 32, lane);
    v16bf b0 = frag_from_lds(sB + (nw2) * 32, 32, lane);
    v16bf b1 = frag_from_lds(sB + (nw2 + 16) * 32, 32, lane);
    acc00 = __builtin_amdgcn_wmma_f32_16x16x32_bf16(false, a0, false, b0, (short)0, acc00, false, false);
    acc01 = __builtin_amdgcn_wmma_f32_16x16x32_bf16(false, a0, false, b1, (short)0, acc01, false, false);
    acc10 = __builtin_amdgcn_wmma_f32_16x16x32_bf16(false, a1, false, b0, (short)0, acc10, false, false);
    acc11 = __builtin_amdgcn_wmma_f32_16x16x32_bf16(false, a1, false, b1, (short)0, acc11, false, false);
    __syncthreads();
  }
  int half = lane >> 4, l15 = lane & 15;
  v8f* accs[2][2] = {{&acc00, &acc01}, {&acc10, &acc11}};
#pragma unroll
  for (int nt = 0; nt < 2; ++nt) {
    int nCol = nBase + nw2 + nt * 16 + l15;
    float bs = bias1 ? bias1[nCol] : 0.f;
    if (bias2) bs += bias2[nCol];
#pragma unroll
    for (int mt = 0; mt < 2; ++mt) {
      v8f a = *accs[mt][nt];
#pragma unroll
      for (int v = 0; v < 8; ++v) {
        int m = mBase + mh2 + mt * 16 + half * 8 + v;
        float val = a[v] + bs;
        if (act == 1) val = tanhf(val);
        size_t orow = trans ? ((size_t)(m % Tdim) * Bdim + (m / Tdim)) : (size_t)m;
        out[orow * (size_t)N + nCol] = val;
      }
    }
  }
}

// ---------------------------------------------------------------------------
// Chunked LSTM recurrence. 32 blocks: blockIdx>>4 = dir, blockIdx&15 = 16-row
// batch tile. W_hh lives in registers as bf16 WMMA B-fragments for the whole
// sequence; h lives in LDS (bf16). The per-step gate-input tile (x@W_ih^T +
// biases, 32KB) is double-buffered into LDS with global_load_async_to_lds_b128
// issued one step ahead; each wave copies/reads only its own 64-column slice,
// so s_wait_asynccnt alone (no barrier) covers the dependency.
// Every 32 steps: h,c <- running w_time-weighted sums (chunk reweighting).
// ---------------------------------------------------------------------------
__global__ __launch_bounds__(256)
void lstm_dir_kernel(const float* __restrict__ xW, const float* __restrict__ Whh,
                     const float* __restrict__ wTime, float* __restrict__ actOut,
                     int T, int Bb) {
  extern __shared__ char smem[];
  unsigned short* sH = (unsigned short*)smem;              // 16x128 bf16   (4KB)
  float* sGate = (float*)(smem + 4096);                    // 16x512 f32    (32KB)
  float* sC    = (float*)(smem + 4096 + 32768);            // 16x128 f32
  float* sHacc = sC + 2048;                                // 16x128 f32
  float* sCacc = sHacc + 2048;                             // 16x128 f32
  float* sX0   = sCacc + 2048;                             // 16x512 f32    (32KB)
  float* sX1   = sX0 + 16 * 512;                           // 16x512 f32    (32KB)

  int tid = threadIdx.x, lane = tid & 31, w = tid >> 5;
  int half = lane >> 4, l15 = lane & 15, lh = lane >> 4;
  int dir = blockIdx.x >> 4;
  int m0 = (blockIdx.x & 15) * 16;
  int nW = w * 64;

  for (int i = tid; i < 2048; i += 256) {
    sH[i] = 0; sC[i] = 0.f; sHacc[i] = 0.f; sCacc[i] = 0.f;
  }

  // Preload W_hh^T fragments: wave w owns gate columns [w*64, w*64+64).
  v16bf wfr[4][4];
#pragma unroll
  for (int nt = 0; nt < 4; ++nt) {
    int n = nW + nt * 16 + l15;
#pragma unroll
    for (int kb = 0; kb < 4; ++kb) {
#pragma unroll
      for (int e = 0; e < 16; ++e) {
        int k = kb * 32 + ((e < 8) ? (half * 8 + e) : (16 + half * 8 + (e - 8)));
        wfr[nt][kb][e] =
            __builtin_bit_cast(__bf16, f2bf_bits(Whh[(size_t)n * 128 + k]));
      }
    }
  }
  __syncthreads();

  // async-copy this wave's 16x64-float slice of the step-t gate-input tile
  auto issue_async = [&](int t_idx, float* sXbuf) {
    const float* gbase = xW + ((size_t)t_idx * Bb + m0) * 512;
#pragma unroll
    for (int q = 0; q < 8; ++q) {
      int row = q * 2 + lh;
      const float* g = gbase + (size_t)row * 512 + nW + l15 * 4;
      unsigned lds = (unsigned)(size_t)(void*)(sXbuf + (size_t)row * 512 + nW + l15 * 4);
      asm volatile("global_load_async_to_lds_b128 %0, %1, off"
                   :: "v"(lds), "v"(g) : "memory");
    }
  };

  issue_async(dir ? (T - 1) : 0, sX0);
  int buf = 0;

  for (int t = 0; t < T; ++t) {
    if (t + 1 < T) {
      issue_async(dir ? (T - 2 - t) : (t + 1), buf ? sX0 : sX1);
      asm volatile("s_wait_asynccnt 0x8" ::: "memory");  // retire older batch only
    } else {
      asm volatile("s_wait_asynccnt 0x0" ::: "memory");
    }
    const float* sXf = buf ? sX1 : sX0;
    // C-init from LDS: precomputed x@W_ih^T + (b_ih+b_hh)
    v8f acc[4];
#pragma unroll
    for (int nt = 0; nt < 4; ++nt) {
      int nCol = nW + nt * 16 + l15;
#pragma unroll
      for (int v = 0; v < 8; ++v)
        acc[nt][v] = sXf[(half * 8 + v) * 512 + nCol];
    }
    // gates += h @ W_hh^T (16 WMMA per wave-step)
#pragma unroll
    for (int kb = 0; kb < 4; ++kb) {
      v16bf a = frag_from_lds(sH + kb * 32, 128, lane);
#pragma unroll
      for (int nt = 0; nt < 4; ++nt)
        acc[nt] = __builtin_amdgcn_wmma_f32_16x16x32_bf16(false, a, false, wfr[nt][kb],
                                                          (short)0, acc[nt], false, false);
    }
#pragma unroll
    for (int nt = 0; nt < 4; ++nt) {
      int nCol = nW + nt * 16 + l15;
#pragma unroll
      for (int v = 0; v < 8; ++v)
        sGate[(half * 8 + v) * 512 + nCol] = acc[nt][v];
    }
    __syncthreads();
    float wt = wTime[t & 31];
    for (int i = tid; i < 2048; i += 256) {
      int m = i >> 7, j = i & 127;
      float gi = sGate[m * 512 + j];
      float gf = sGate[m * 512 + 128 + j];
      float gg = sGate[m * 512 + 256 + j];
      float go = sGate[m * 512 + 384 + j];
      float cn = sigm(gf) * sC[i] + sigm(gi) * tanhf(gg);
      float hn = sigm(go) * tanhf(cn);
      sC[i] = cn;
      sH[i] = f2bf_bits(hn);
      sHacc[i] += wt * hn;
      sCacc[i] += wt * cn;
      actOut[((size_t)t * Bb + m0 + m) * 256 + dir * 128 + j] = hn;
    }
    __syncthreads();
    if ((t & 31) == 31) {  // chunk boundary: reweighted carry + overwrite last output
      for (int i = tid; i < 2048; i += 256) {
        int m = i >> 7, j = i & 127;
        float hn = sHacc[i], cn = sCacc[i];
        sH[i] = f2bf_bits(hn);
        sC[i] = cn;
        sHacc[i] = 0.f;
        sCacc[i] = 0.f;
        actOut[((size_t)t * Bb + m0 + m) * 256 + dir * 128 + j] = hn;
      }
      __syncthreads();
    }
    buf ^= 1;
  }
}

// ---------------------------------------------------------------------------
// Attention tail (bandwidth-trivial VALU kernels, deterministic reductions)
// ---------------------------------------------------------------------------
__global__ __launch_bounds__(256)
void pot_kernel(const float* __restrict__ th, const float* __restrict__ Wa2,
                float* __restrict__ pot, int TB) {
  int row = blockIdx.x * blockDim.x + threadIdx.x;
  if (row >= TB) return;
  const float* tr = th + (size_t)row * 128;
#pragma unroll
  for (int r = 0; r < 8; ++r) {
    float s = 0.f;
    for (int d = 0; d < 128; ++d) s += tr[d] * Wa2[r * 128 + d];
    pot[(size_t)row * 8 + r] = s;
  }
}

__global__ __launch_bounds__(128)
void softmax_kernel(const float* __restrict__ pot, const int* __restrict__ length,
                    float* __restrict__ wght, int T, int Bb) {
  int b = blockIdx.x >> 3, r = blockIdx.x & 7;
  int len = length[b];
  int tid = threadIdx.x;
  __shared__ float red[128];
  float mx = -3.4e38f;
  for (int t = tid; t < T; t += 128)
    if (t < len) mx = fmaxf(mx, pot[((size_t)t * Bb + b) * 8 + r]);
  red[tid] = mx; __syncthreads();
  for (int s = 64; s > 0; s >>= 1) {
    if (tid < s) red[tid] = fmaxf(red[tid], red[tid + s]);
    __syncthreads();
  }
  mx = red[0]; __syncthreads();
  float sum = 0.f;
  for (int t = tid; t < T; t += 128)
    if (t < len) sum += __expf(pot[((size_t)t * Bb + b) * 8 + r] - mx);
  red[tid] = sum; __syncthreads();
  for (int s = 64; s > 0; s >>= 1) {
    if (tid < s) red[tid] += red[tid + s];
    __syncthreads();
  }
  float inv = 1.0f / red[0];
  for (int t = tid; t < T; t += 128) {
    float v = (t < len) ? __expf(pot[((size_t)t * Bb + b) * 8 + r] - mx) * inv : 0.f;
    wght[((size_t)b * 8 + r) * T + t] = v;
  }
}

__global__ __launch_bounds__(256)
void pooled_kernel(const float* __restrict__ wght, const float* __restrict__ act,
                   float* __restrict__ pooled, int T, int Bb) {
  int b = blockIdx.x, tid = threadIdx.x;
  for (int idx = tid; idx < 2048; idx += 256) {
    int r = idx >> 8, h = idx & 255;
    const float* wr = wght + ((size_t)b * 8 + r) * T;
    float s = 0.f;
    for (int t = 0; t < T; ++t) s += wr[t] * act[((size_t)t * Bb + b) * 256 + h];
    pooled[(size_t)b * 2048 + idx] = s;
  }
}

__global__ __launch_bounds__(256)
void final_kernel(const float* __restrict__ pooled, const float* __restrict__ Wout,
                  const float* __restrict__ bout, float* __restrict__ out, int Bb) {
  int b = blockIdx.x, tid = threadIdx.x;
  __shared__ float red[256];
  __shared__ float logits[7];
  for (int o = 0; o < 7; ++o) {
    float s = 0.f;
    for (int j = tid; j < 2048; j += 256)
      s += pooled[(size_t)b * 2048 + j] * Wout[(size_t)o * 2048 + j];
    red[tid] = s; __syncthreads();
    for (int st = 128; st > 0; st >>= 1) {
      if (tid < st) red[tid] += red[tid + st];
      __syncthreads();
    }
    if (tid == 0) logits[o] = red[0] + bout[o];
    __syncthreads();
  }
  if (tid == 0) {
    float mx = logits[0];
    for (int o = 1; o < 7; ++o) mx = fmaxf(mx, logits[o]);
    float sum = 0.f;
    for (int o = 0; o < 7; ++o) sum += __expf(logits[o] - mx);
    float lse = mx + __logf(sum);
    for (int o = 0; o < 7; ++o) out[(size_t)b * 7 + o] = logits[o] - lse;
  }
}

__global__ __launch_bounds__(64)
void gram_kernel(const float* __restrict__ wght, float* __restrict__ penp, int T) {
  int b = blockIdx.x, tid = threadIdx.x;
  int r = tid >> 3, s = tid & 7;
  const float* wr = wght + ((size_t)b * 8 + r) * T;
  const float* wsp = wght + ((size_t)b * 8 + s) * T;
  float g = 0.f;
  for (int t = 0; t < T; ++t) g += sqrtf(wr[t] * wsp[t]);  // sqrt(w)>=0
  float d = g - 1.0f;
  __shared__ float red[64];
  red[tid] = d * d; __syncthreads();
  for (int st = 32; st > 0; st >>= 1) {
    if (tid < st) red[tid] += red[tid + st];
    __syncthreads();
  }
  if (tid == 0) penp[b] = red[0];
}

__global__ __launch_bounds__(256)
void penalty_reduce(const float* __restrict__ penp, float* __restrict__ out, int Bb) {
  __shared__ float red[256];
  int tid = threadIdx.x;
  red[tid] = (tid < Bb) ? penp[tid] : 0.f;
  __syncthreads();
  for (int st = 128; st > 0; st >>= 1) {
    if (tid < st) red[tid] += red[tid + st];
    __syncthreads();
  }
  if (tid == 0) out[0] = red[0];
}

extern "C" void kernel_launch(void* const* d_in, const int* in_sizes, int n_in,
                              void* d_out, int out_size, void* d_ws, size_t ws_size,
                              hipStream_t stream) {
  (void)in_sizes; (void)n_in; (void)out_size; (void)ws_size;
  const float* x      = (const float*)d_in[0];
  const float* W_in   = (const float*)d_in[1];
  const float* b_in   = (const float*)d_in[2];
  const float* W_ih   = (const float*)d_in[3];
  const float* b_ih   = (const float*)d_in[4];
  const float* W_hh   = (const float*)d_in[5];
  const float* b_hh   = (const float*)d_in[6];
  const float* w_time = (const float*)d_in[7];
  const float* W_a1   = (const float*)d_in[8];
  const float* b_a1   = (const float*)d_in[9];
  const float* W_a2   = (const float*)d_in[10];
  const float* W_out  = (const float*)d_in[11];
  const float* b_out  = (const float*)d_in[12];
  const int*   length = (const int*)d_in[13];

  const int Bb = 256, T = 512, TB = Bb * T;
  char* ws = (char*)d_ws;
  float* xW   = (float*)ws;                            // TB*512 f32 (268MB)
  float* actA = (float*)(ws + (size_t)TB * 512 * 4);   // TB*256 f32
  float* actB = actA + (size_t)TB * 256;               // TB*256 f32
  // attention scratch reuses the xW region (layers done by then)
  float* th     = (float*)ws;                          // TB*128
  float* pot    = th + (size_t)TB * 128;               // TB*8
  float* wght   = pot + (size_t)TB * 8;                // Bb*8*T
  float* pooled = wght + (size_t)Bb * 8 * T;           // Bb*2048
  float* penp   = pooled + (size_t)Bb * 2048;          // Bb
  float* outF = (float*)d_out;

  const size_t lstm_lds = 4096 + 32768 + 3 * 8192 + 2 * 32768;  // 124KB dynamic LDS

  // 1) input projection, written transposed [B,T]->[T,B] for the recurrence
  gemm_bf16_wmma<<<dim3(256 / 128, TB / 64), 256, 0, stream>>>(
      x, W_in, b_in, nullptr, actA, TB, 128, 256, 0, 1, T, Bb);

  // 2) two bidirectional chunked-LSTM layers (shared weights)
  float* actIn = actA;
  float* actNxt = actB;
  for (int layer = 0; layer < 2; ++layer) {
    gemm_bf16_wmma<<<dim3(512 / 128, TB / 64), 256, 0, stream>>>(
        actIn, W_ih, b_ih, b_hh, xW, TB, 256, 512, 0, 0, T, Bb);
    lstm_dir_kernel<<<32, 256, lstm_lds, stream>>>(xW, W_hh, w_time, actNxt, T, Bb);
    float* tmp = actIn; actIn = actNxt; actNxt = tmp;
  }
  // after 2 layers actIn == actA (outside the reused xW region)

  // 3) attention + heads
  gemm_bf16_wmma<<<dim3(128 / 128, TB / 64), 256, 0, stream>>>(
      actIn, W_a1, b_a1, nullptr, th, TB, 256, 128, 1, 0, T, Bb);
  pot_kernel<<<(TB + 255) / 256, 256, 0, stream>>>(th, W_a2, pot, TB);
  softmax_kernel<<<Bb * 8, 128, 0, stream>>>(pot, length, wght, T, Bb);
  pooled_kernel<<<Bb, 256, 0, stream>>>(wght, actIn, pooled, T, Bb);
  final_kernel<<<Bb, 256, 0, stream>>>(pooled, W_out, b_out, outF, Bb);
  gram_kernel<<<Bb, 64, 0, stream>>>(wght, penp, T);
  penalty_reduce<<<1, 256, 0, stream>>>(penp, outF + (size_t)Bb * 7, Bb);
}
